// StructuredMaskedAttention_26173530701940
// MI455X (gfx1250) — matrix operand order repaired and checked by reading, not compiled, not run on hardware
//
#include <hip/hip_runtime.h>

// ---------------------------------------------------------------------------
// StructuredMaskedAttention fused for MI455X (gfx1250, wave32, WMMA).
//
// Math folding:
//   Y   = X @ G          with G[k, h*56+o] = sum_dd B[h,dd,k] * E_h[dd,o],
//                             E_h = (suffix-sum-rows A_h) @ C_h^T
//   out = rsig * (Y @ W'^T - mu * s1) + c0
//         W' = gamma ⊙ W ; s1[o] = Σ_i gamma_i W[o,i] ; c0[o] = Σ_i beta_i W[o,i] + bias[o]
//
// GEMMs run on v_wmma_f32_16x16x32_bf16 with a bf16 hi/lo error-compensated
// split (3 products) to retain ~fp32 accuracy. B-operands and output use
// uniform-base + 32-bit-offset addressing so the backend emits saddr-form
// global ops with folded immediates instead of per-lane 64-bit adds.
// ---------------------------------------------------------------------------

typedef __attribute__((ext_vector_type(16))) __bf16        bf16x16;
typedef __attribute__((ext_vector_type(4)))  __bf16        bf16x4;
typedef __attribute__((ext_vector_type(8)))  float         f32x8;
typedef __attribute__((ext_vector_type(4)))  unsigned int  u32x4;
typedef __attribute__((ext_vector_type(2)))  unsigned int  u32x2;

struct Q32 { u32x4 a, b; };

__device__ __forceinline__ bf16x16 pack_ab(u32x4 a, u32x4 b) {
    Q32 q; q.a = a; q.b = b;
    return __builtin_bit_cast(bf16x16, q);
}

__device__ __forceinline__ void split_bf16(float v, __bf16* hp, __bf16* lp) {
    __bf16 h = (__bf16)v;
    *hp = h;
    *lp = (__bf16)(v - (float)h);
}

#define DIN   448
#define ROWS  32              // rows of X per workgroup
#define LSTR  456             // padded LDS row stride (bf16); 912B, 16B aligned
#define NT    14              // 14 N-tiles of 16 per wave (half of 448)
#define TILEB (16 * DIN * 2)  // byte stride between consecutive 16-col tiles (14336)

__device__ __forceinline__ f32x8 wmma_bf16(bf16x16 a, bf16x16 b, f32x8 c) {
    return __builtin_amdgcn_wmma_f32_16x16x32_bf16(false, a, false, b,
                                                   (short)0, c, false, false);
}

// A-operand (16x32 bf16), contiguous K to match B: lane 0-15 row M=lane holds K k0+0..15,
// lane 16-31 row M=lane-16 holds K k0+16..31.
__device__ __forceinline__ bf16x16 load_a(const __bf16* rowp, int k0, int hl) {
    const u32x4* p1 = (const u32x4*)(rowp + k0 + 16 * hl);
    const u32x4* p2 = (const u32x4*)(rowp + k0 + 8 + 16 * hl);
    return pack_ab(*p1, *p2);
}

// B-operand (32x16 bf16): lane 0-15 -> N=lane, K=k0+0..15 ; lane 16-31 -> N=lane-16,
// K=k0+16..31. Matrix stored row-major as Mat[n][k] -> 32B contiguous per lane.
// Uniform base + 32-bit byte offset so backend emits saddr-form global_load_b128.
__device__ __forceinline__ bf16x16 load_bg(const __bf16* __restrict__ mat,
                                           unsigned byteOff) {
    const char* p = (const char*)mat;
    u32x4 lo = *(const u32x4*)(p + byteOff);
    u32x4 hi = *(const u32x4*)(p + byteOff + 16);
    return pack_ab(lo, hi);
}

// Split a float4 into bf16 hi/lo packed as u32x2 each (for ds_store_b64).
__device__ __forceinline__ void split4(float4 v, u32x2* hq, u32x2* lq) {
    bf16x4 h, l;
    h[0] = (__bf16)v.x;  l[0] = (__bf16)(v.x - (float)h[0]);
    h[1] = (__bf16)v.y;  l[1] = (__bf16)(v.y - (float)h[1]);
    h[2] = (__bf16)v.z;  l[2] = (__bf16)(v.z - (float)h[2]);
    h[3] = (__bf16)v.w;  l[3] = (__bf16)(v.w - (float)h[3]);
    *hq = __builtin_bit_cast(u32x2, h);
    *lq = __builtin_bit_cast(u32x2, l);
}

// ---------------------------------------------------------------------------
// Prep kernel: blocks 0..7 build G^T (bf16 hi/lo); blocks 8..455 fold W.
// ---------------------------------------------------------------------------
__global__ __launch_bounds__(256) void prep_kernel(
    const float* __restrict__ A, const float* __restrict__ B,
    const float* __restrict__ C, const float* __restrict__ gamma,
    const float* __restrict__ beta, const float* __restrict__ W,
    const float* __restrict__ bias,
    __bf16* __restrict__ GhT, __bf16* __restrict__ GlT,
    __bf16* __restrict__ Wh,  __bf16* __restrict__ Wl,
    float* __restrict__ s1g,  float* __restrict__ c0g)
{
    __shared__ float S[56][57];
    __shared__ float E[56][57];
    __shared__ float red1[256];
    __shared__ float red2[256];
    const int tid = threadIdx.x;

    if (blockIdx.x < 8) {
        const int h = blockIdx.x;
        const float* Ah = A + h * 56 * 56;
        const float* Bh = B + h * 56 * 448;
        const float* Ch = C + h * 56 * 56;
        // S = A_h @ M  (row-wise suffix sums)
        if (tid < 56) {
            float run = 0.f;
            for (int d2 = 55; d2 >= 0; --d2) {
                run += Ah[tid * 56 + d2];
                S[tid][d2] = run;
            }
        }
        __syncthreads();
        // E[dd][o] = Σ_e S[dd][e] * C[o][e]
        for (int idx = tid; idx < 56 * 56; idx += 256) {
            int dd = idx / 56, o = idx % 56;
            float acc = 0.f;
            for (int e = 0; e < 56; ++e) acc += S[dd][e] * Ch[o * 56 + e];
            E[dd][o] = acc;
        }
        __syncthreads();
        // G^T[n=h*56+o][k] = Σ_dd B[dd][k] * E[dd][o], split into bf16 hi/lo
        for (int idx = tid; idx < 56 * 448; idx += 256) {
            int o = idx / 448, k = idx % 448;
            float acc = 0.f;
            for (int dd = 0; dd < 56; ++dd) acc += Bh[dd * 448 + k] * E[dd][o];
            size_t n = (size_t)(h * 56 + o) * 448 + k;
            split_bf16(acc, &GhT[n], &GlT[n]);
        }
    } else {
        const int o = blockIdx.x - 8;   // 0..447
        float a1 = 0.f, a2 = 0.f;
        for (int i = tid; i < 448; i += 256) {
            float w  = W[o * 448 + i];
            float wp = gamma[i] * w;
            split_bf16(wp, &Wh[o * 448 + i], &Wl[o * 448 + i]);
            a1 += wp;
            a2 += beta[i] * w;
        }
        red1[tid] = a1; red2[tid] = a2;
        __syncthreads();
        for (int s = 128; s >= 1; s >>= 1) {
            if (tid < s) { red1[tid] += red1[tid + s]; red2[tid] += red2[tid + s]; }
            __syncthreads();
        }
        if (tid == 0) { s1g[o] = red1[0]; c0g[o] = red2[0] + bias[o]; }
    }
}

// ---------------------------------------------------------------------------
// Fused main kernel: X-load/split -> WMMA GEMM1 -> LN stats -> WMMA GEMM2 ->
// LN fix-up epilogue. 128 threads = 4 waves; wave w: M-tile (w&1), N-half (w>>1).
// ---------------------------------------------------------------------------
__global__ __launch_bounds__(128) void fused_kernel(
    const float* __restrict__ x,
    const __bf16* __restrict__ GhT, const __bf16* __restrict__ GlT,
    const __bf16* __restrict__ Wh,  const __bf16* __restrict__ Wl,
    const float* __restrict__ s1g,  const float* __restrict__ c0g,
    float* __restrict__ out)
{
    __shared__ __bf16 Xh[ROWS][LSTR];
    __shared__ __bf16 Xl[ROWS][LSTR];
    __shared__ float  sumY[ROWS], sumY2[ROWS], muL[ROWS], rsL[ROWS];

    const int tid    = threadIdx.x;
    const int lane   = tid & 31;
    const int wave   = tid >> 5;
    const int laneLo = lane & 15;
    const int hl     = lane >> 4;
    const int mt     = wave & 1;    // which 16-row tile
    const int ng     = wave >> 1;   // which 224-col half
    const size_t R0  = (size_t)blockIdx.x * ROWS;

    if (tid < ROWS) { sumY[tid] = 0.f; sumY2[tid] = 0.f; }

    // Stage 1: load X block, split to bf16 hi/lo, store packed b64 to LDS
    {
        const float4* xb = (const float4*)(x + R0 * DIN);
        for (int i = tid; i < ROWS * (DIN / 4); i += 128) {
            float4 v = xb[i];
            int row = i / (DIN / 4);
            int c   = (i % (DIN / 4)) * 4;
            u32x2 hq, lq;
            split4(v, &hq, &lq);
            *(u32x2*)&Xh[row][c] = hq;
            *(u32x2*)&Xl[row][c] = lq;
        }
    }
    __syncthreads();

    const int rowA = mt * 16 + laneLo;
    const __bf16* arow_h = &Xh[rowA][0];
    const __bf16* arow_l = &Xl[rowA][0];
    const int nbase = ng * 224 + laneLo;
    // per-lane byte offset of this lane's first B tile row (+16*hl K elems)
    const unsigned bcol = ((unsigned)nbase * DIN + 16u * hl) * 2u;

    // Stage 2: GEMM1  Y = X @ G  (bf16x3 split)
    f32x8 acc[NT];
#pragma unroll
    for (int t = 0; t < NT; ++t)
#pragma unroll
        for (int j = 0; j < 8; ++j) acc[t][j] = 0.f;

#pragma unroll 2
    for (int k0 = 0; k0 < DIN; k0 += 32) {
        bf16x16 ah = load_a(arow_h, k0, hl);
        bf16x16 al = load_a(arow_l, k0, hl);
        const unsigned bo = bcol + (unsigned)k0 * 2u;
#pragma unroll
        for (int t = 0; t < NT; ++t) {
            bf16x16 bh = load_bg(GhT, bo + (unsigned)(t * TILEB));
            bf16x16 bl = load_bg(GlT, bo + (unsigned)(t * TILEB));
            acc[t] = wmma_bf16(ah, bh, acc[t]);
            acc[t] = wmma_bf16(ah, bl, acc[t]);
            acc[t] = wmma_bf16(al, bh, acc[t]);
        }
    }
    __syncthreads();   // everyone finished reading X from LDS

    // Stage 3: write raw Y (bf16 hi/lo) over X in LDS; accumulate LN stats
    {
        float s[8], s2[8];
#pragma unroll
        for (int r = 0; r < 8; ++r) { s[r] = 0.f; s2[r] = 0.f; }
#pragma unroll
        for (int t = 0; t < NT; ++t) {
            int col = ng * 224 + t * 16 + laneLo;
#pragma unroll
            for (int r = 0; r < 8; ++r) {
                float v = acc[t][r];
                s[r]  += v;
                s2[r] += v * v;
                int row = mt * 16 + r + 8 * hl;
                split_bf16(v, &Xh[row][col], &Xl[row][col]);
            }
        }
#pragma unroll
        for (int r = 0; r < 8; ++r) {
#pragma unroll
            for (int m = 8; m >= 1; m >>= 1) {
                s[r]  += __shfl_xor(s[r],  m, 32);
                s2[r] += __shfl_xor(s2[r], m, 32);
            }
            if (laneLo == 0) {
                int row = mt * 16 + r + 8 * hl;
                atomicAdd(&sumY[row],  s[r]);
                atomicAdd(&sumY2[row], s2[r]);
            }
        }
    }
    __syncthreads();
    if (tid < ROWS) {
        float mu  = sumY[tid] * (1.f / DIN);
        float var = sumY2[tid] * (1.f / DIN) - mu * mu;
        muL[tid] = mu;
        rsL[tid] = rsqrtf(var + 1e-5f);
    }
    __syncthreads();

    // Stage 4: GEMM2  Z = Y @ W'^T  (bf16x3 split)
#pragma unroll
    for (int t = 0; t < NT; ++t)
#pragma unroll
        for (int j = 0; j < 8; ++j) acc[t][j] = 0.f;

#pragma unroll 2
    for (int k0 = 0; k0 < DIN; k0 += 32) {
        bf16x16 ah = load_a(arow_h, k0, hl);   // now holds Y hi
        bf16x16 al = load_a(arow_l, k0, hl);   // Y lo
        const unsigned bo = bcol + (unsigned)k0 * 2u;
#pragma unroll
        for (int t = 0; t < NT; ++t) {
            bf16x16 bh = load_bg(Wh, bo + (unsigned)(t * TILEB));
            bf16x16 bl = load_bg(Wl, bo + (unsigned)(t * TILEB));
            acc[t] = wmma_bf16(ah, bh, acc[t]);
            acc[t] = wmma_bf16(ah, bl, acc[t]);
            acc[t] = wmma_bf16(al, bh, acc[t]);
        }
    }

    // Stage 5: LN fix-up epilogue:  out = rsig*(Z - mu*s1) + c0
    float muv[8], rsv[8];
#pragma unroll
    for (int r = 0; r < 8; ++r) {
        int row = mt * 16 + r + 8 * hl;
        muv[r] = muL[row];
        rsv[r] = rsL[row];
    }
    const char* s1b = (const char*)s1g;
    const char* c0b = (const char*)c0g;
    char* ob = (char*)out;
    // uniform base + u32 per-lane offset; tile/row deltas are constants
    const unsigned ocol  = (unsigned)(ng * 224 + laneLo);
    const unsigned obase = ((unsigned)(R0 + (size_t)(mt * 16 + 8 * hl)) * DIN + ocol) * 4u;
#pragma unroll
    for (int t = 0; t < NT; ++t) {
        float s1v = *(const float*)(s1b + ocol * 4u + (unsigned)(t * 64));
        float c0v = *(const float*)(c0b + ocol * 4u + (unsigned)(t * 64));
#pragma unroll
        for (int r = 0; r < 8; ++r) {
            float v = rsv[r] * (acc[t][r] - muv[r] * s1v) + c0v;
            *(float*)(ob + obase + (unsigned)(t * 64 + r * DIN * 4)) = v;
        }
    }
}

// ---------------------------------------------------------------------------
extern "C" void kernel_launch(void* const* d_in, const int* in_sizes, int n_in,
                              void* d_out, int out_size, void* d_ws, size_t ws_size,
                              hipStream_t stream) {
    const float* x     = (const float*)d_in[0];
    const float* A     = (const float*)d_in[1];
    const float* B     = (const float*)d_in[2];
    const float* C     = (const float*)d_in[3];
    const float* gamma = (const float*)d_in[4];
    const float* beta  = (const float*)d_in[5];
    const float* W     = (const float*)d_in[6];
    const float* bias  = (const float*)d_in[7];
    float* out = (float*)d_out;

    char* p = (char*)d_ws;
    __bf16* GhT = (__bf16*)p; p += (size_t)DIN * DIN * sizeof(__bf16);
    __bf16* GlT = (__bf16*)p; p += (size_t)DIN * DIN * sizeof(__bf16);
    __bf16* Whp = (__bf16*)p; p += (size_t)DIN * DIN * sizeof(__bf16);
    __bf16* Wlp = (__bf16*)p; p += (size_t)DIN * DIN * sizeof(__bf16);
    float*  s1g = (float*)p;  p += DIN * sizeof(float);
    float*  c0g = (float*)p;  p += DIN * sizeof(float);

    // Fold the whole structured-attention weight chain + LN/W folding.
    prep_kernel<<<8 + DIN, 256, 0, stream>>>(A, B, C, gamma, beta, W, bias,
                                             GhT, GlT, Whp, Wlp, s1g, c0g);

    const int rows = in_sizes[0] / DIN;   // 65536
    fused_kernel<<<rows / ROWS, 128, 0, stream>>>(x, GhT, GlT, Whp, Wlp,
                                                  s1g, c0g, out);
}
